// UpDownCaptionerText_44933947851191
// MI455X (gfx1250) — matrix operand order repaired and checked by reading, not compiled, not run on hardware
//
#include <hip/hip_runtime.h>
#include <hip/hip_bf16.h>
#include <math.h>

// Problem dims (fixed by the reference)
#define B_ 128
#define R_ 36
#define F_ 2048
#define E_ 512
#define H_ 512
#define A_ 512
#define V_ 32000
#define T_ 20

typedef __attribute__((ext_vector_type(16))) __bf16 v16bf;
typedef __attribute__((ext_vector_type(8)))  __bf16 v8bf;
typedef __attribute__((ext_vector_type(8)))  float  v8f;
typedef __attribute__((ext_vector_type(4)))  float  v4f;

// f32 -> bf16 round-to-nearest-even (never used in GEMM inner loops)
static __device__ __forceinline__ __bf16 f2bf(float f) {
  union { float f; unsigned int u; } v; v.f = f;
  unsigned int r = v.u + 0x7FFFu + ((v.u >> 16) & 1u);
  unsigned short h = (unsigned short)(r >> 16);
  return __builtin_bit_cast(__bf16, h);
}

// ---------------------------------------------------------------------------
// bf16 GEMM with optional second K-segment (register-fused accumulate):
//   C[M,N] = bias[N] + Xa[M,Ka] @ Wa[N,Ka]^T  (+ Xb[M,Kb] @ Wb[N,Kb]^T if Xb)
// One wave computes a 32x64 f32 tile (2 row-tiles x 4 col-tiles). Per 32-wide
// K slice: 12 b128 loads (2 A fragments + 4 B fragments) then 8 back-to-back
// WMMAs; K loop unrolled 2x so next-slice loads issue under current WMMAs.
//
// Leading dimensions are compile-time (LD1 for segment 1; segment 2 is always
// H=512): all fragment addresses become immediate offsets off one advancing
// base pointer per matrix, freeing the address VGPRs that previously spilled.
//
// VGPR layouts used here (wave32; A/B use contiguous k order, not the ISA table order):
//   A (16x32 bf16):  lane m = L%16; one contiguous 32B run at k0+hi*16
//   B (32x16 bf16):  lane n = L%16; one contiguous 32B run at k0+hi*16
//   C/D (16x16 f32): lane n = L%16; vgpr j -> M = j + 8*(L>=16)
// ---------------------------------------------------------------------------
template<int MT, int LD1>
__global__ void gemm_wmma_bf16(const __bf16* __restrict__ Xa,
                               const __bf16* __restrict__ Wa, int Ka,
                               const __bf16* __restrict__ Xb,
                               const __bf16* __restrict__ Wb, int Kb,
                               const float* __restrict__ bias,
                               float* __restrict__ C, int ldc,
                               int M, int N)
{
  const int lane = threadIdx.x & 31;
  const int wv   = threadIdx.x >> 5;
  const int n64  = N >> 6;                        // col tiles of 64
  const int gw   = blockIdx.x * (blockDim.x >> 5) + wv;
  const int total = (M / (16 * MT)) * n64;
  if (gw >= total) return;                        // wave-uniform exit (EXEC all-1)

  const int row0 = (gw / n64) * (16 * MT);
  const int col0 = (gw % n64) << 6;
  const int m  = lane & 15;
  const int n  = lane & 15;
  const int hi = lane >> 4;

  v8f acc[MT][4];
#pragma unroll
  for (int t = 0; t < 4; ++t) {
    float bv = bias ? bias[col0 + 16 * t + n] : 0.0f;
#pragma unroll
    for (int r = 0; r < MT; ++r)
#pragma unroll
      for (int j = 0; j < 8; ++j) acc[r][t][j] = bv;
  }

  // ---- segment 1 (leading dim LD1, compile-time) ----
  {
    const __bf16* xbase = Xa + (long)(row0 + m) * LD1 + hi * 16;
    const __bf16* wbase = Wa + (long)(col0 + n) * LD1 + hi * 16;
#pragma unroll 2
    for (int k0 = 0; k0 < Ka; k0 += 32) {
      v16bf b[4];
#pragma unroll
      for (int t = 0; t < 4; ++t)
        b[t] = *(const v16bf*)(wbase + 16 * t * LD1 + k0);   // imm offsets
      v16bf a[MT];
#pragma unroll
      for (int r = 0; r < MT; ++r) {
        v8bf lo  = *(const v8bf*)(xbase + 16 * r * LD1 + k0);
        v8bf hi8 = *(const v8bf*)(xbase + 16 * r * LD1 + k0 + 8);
        a[r] = __builtin_shufflevector(lo, hi8, 0, 1, 2, 3, 4, 5, 6, 7,
                                       8, 9, 10, 11, 12, 13, 14, 15);
      }
#pragma unroll
      for (int t = 0; t < 4; ++t)
#pragma unroll
        for (int r = 0; r < MT; ++r)
          acc[r][t] = __builtin_amdgcn_wmma_f32_16x16x32_bf16(
              false, a[r], false, b[t], (short)0, acc[r][t], false, false);
    }
  }

  // ---- optional segment 2 (leading dim H_, compile-time) ----
  if (Xb) {
    const __bf16* xbase = Xb + (long)(row0 + m) * H_ + hi * 16;
    const __bf16* wbase = Wb + (long)(col0 + n) * H_ + hi * 16;
#pragma unroll 2
    for (int k0 = 0; k0 < Kb; k0 += 32) {
      v16bf b[4];
#pragma unroll
      for (int t = 0; t < 4; ++t)
        b[t] = *(const v16bf*)(wbase + 16 * t * H_ + k0);
      v16bf a[MT];
#pragma unroll
      for (int r = 0; r < MT; ++r) {
        v8bf lo  = *(const v8bf*)(xbase + 16 * r * H_ + k0);
        v8bf hi8 = *(const v8bf*)(xbase + 16 * r * H_ + k0 + 8);
        a[r] = __builtin_shufflevector(lo, hi8, 0, 1, 2, 3, 4, 5, 6, 7,
                                       8, 9, 10, 11, 12, 13, 14, 15);
      }
#pragma unroll
      for (int t = 0; t < 4; ++t)
#pragma unroll
        for (int r = 0; r < MT; ++r)
          acc[r][t] = __builtin_amdgcn_wmma_f32_16x16x32_bf16(
              false, a[r], false, b[t], (short)0, acc[r][t], false, false);
    }
  }

#pragma unroll
  for (int r = 0; r < MT; ++r)
#pragma unroll
    for (int t = 0; t < 4; ++t)
#pragma unroll
      for (int j = 0; j < 8; ++j)
        C[(long)(row0 + 16 * r + j + hi * 8) * ldc + col0 + 16 * t + n] =
            acc[r][t][j];
}

// ---------------------------------------------------------------------------
// Vectorized f32 -> bf16 (8 elements/thread; n multiple of 8 — true here).
// ---------------------------------------------------------------------------
__global__ void f32_to_bf16_v8(const float* __restrict__ src,
                               __bf16* __restrict__ dst, int n8)
{
  int idx = blockIdx.x * blockDim.x + threadIdx.x;
  if (idx >= n8) return;
  v4f x0 = *(const v4f*)(src + (long)idx * 8);
  v4f x1 = *(const v4f*)(src + (long)idx * 8 + 4);
  v8bf y;
#pragma unroll
  for (int e = 0; e < 4; ++e) y[e] = f2bf(x0[e]);
#pragma unroll
  for (int e = 0; e < 4; ++e) y[4 + e] = f2bf(x1[e]);
  *(v8bf*)(dst + (long)idx * 8) = y;
}

__global__ void zero_f32(float* __restrict__ p, int n)
{
  int idx = blockIdx.x * blockDim.x + threadIdx.x;
  if (idx < n) p[idx] = 0.f;
}

__global__ void zero_bf16(__bf16* __restrict__ p, int n)
{
  int idx = blockIdx.x * blockDim.x + threadIdx.x;
  if (idx < n) p[idx] = __builtin_bit_cast(__bf16, (unsigned short)0);
}

// ---------------------------------------------------------------------------
// Attention: e[b,r] = relu(att1[b,r,:] + att2[b,:]) . Wa + ba ; softmax over R;
// context[b,:] = sum_r alpha[r] * features[b,r,:].  One block per batch row.
// ---------------------------------------------------------------------------
__global__ void attention_kernel(const float* __restrict__ att1,     // [B*R, A]
                                 const float* __restrict__ att2,     // [B, A]
                                 const float* __restrict__ Wa,       // [A]
                                 const float* __restrict__ ba,       // [1]
                                 const float* __restrict__ features, // [B*R, F]
                                 float* __restrict__ context)        // [B, F]
{
  const int b = blockIdx.x;
  __shared__ float esh[R_];
  const int lane = threadIdx.x & 31;
  const int wv   = threadIdx.x >> 5;
  const int nw   = blockDim.x >> 5;

  for (int r = wv; r < R_; r += nw) {
    float s = 0.f;
    const float* a1 = att1 + (long)(b * R_ + r) * A_;
    const float* a2 = att2 + (long)b * A_;
    for (int a = lane; a < A_; a += 32) {
      float v = a1[a] + a2[a];
      v = v > 0.f ? v : 0.f;
      s += v * Wa[a];
    }
#pragma unroll
    for (int off = 16; off > 0; off >>= 1) s += __shfl_down(s, off, 32);
    if (lane == 0) esh[r] = s + ba[0];
  }
  __syncthreads();

  if (threadIdx.x == 0) {
    float mx = esh[0];
    for (int r = 1; r < R_; ++r) mx = fmaxf(mx, esh[r]);
    float sum = 0.f;
    for (int r = 0; r < R_; ++r) { float e = __expf(esh[r] - mx); esh[r] = e; sum += e; }
    float inv = 1.f / sum;
    for (int r = 0; r < R_; ++r) esh[r] *= inv;
  }
  __syncthreads();

  for (int f = threadIdx.x; f < F_; f += blockDim.x) {
    float acc = 0.f;
#pragma unroll 4
    for (int r = 0; r < R_; ++r)
      acc += esh[r] * features[(long)(b * R_ + r) * F_ + f];
    context[(long)b * F_ + f] = acc;
  }
}

// x1 = [h_lang(H) | context(F) | embedding[captions[:,t]](E)]  -> bf16
__global__ void build_x1(const float* __restrict__ h_lang,
                         const float* __restrict__ context,
                         const float* __restrict__ embedding,
                         const int* __restrict__ captions, int t,
                         __bf16* __restrict__ x1)
{
  const int X1 = H_ + F_ + E_;
  int idx = blockIdx.x * blockDim.x + threadIdx.x;
  if (idx >= B_ * X1) return;
  int b = idx / X1, j = idx - b * X1;
  float v;
  if (j < H_)            v = h_lang[b * H_ + j];
  else if (j < H_ + F_)  v = context[(long)b * F_ + (j - H_)];
  else {
    int w = captions[b * T_ + t];
    v = embedding[(long)w * E_ + (j - H_ - F_)];
  }
  x1[(long)b * X1 + j] = f2bf(v);
}

// x2 = [context(F) | h_att(H)]  -> bf16
__global__ void build_x2(const float* __restrict__ context,
                         const float* __restrict__ h_att,
                         __bf16* __restrict__ x2)
{
  const int X2 = F_ + H_;
  int idx = blockIdx.x * blockDim.x + threadIdx.x;
  if (idx >= B_ * X2) return;
  int b = idx / X2, j = idx - b * X2;
  float v = (j < F_) ? context[(long)b * F_ + j] : h_att[b * H_ + (j - F_)];
  x2[(long)b * X2 + j] = f2bf(v);
}

// LSTM pointwise update: gates [B, 4H] laid out [i | f | g | o].
// Writes f32 state plus a bf16 mirror of h for downstream GEMMs.
__global__ void lstm_pointwise(const float* __restrict__ gates,
                               const float* __restrict__ bhh,
                               float* __restrict__ h, float* __restrict__ c,
                               __bf16* __restrict__ hb)
{
  int idx = blockIdx.x * blockDim.x + threadIdx.x;
  if (idx >= B_ * H_) return;
  int b = idx / H_, u = idx - b * H_;
  const float* g = gates + (long)b * 4 * H_;
  float gi = g[u]          + bhh[u];
  float gf = g[H_ + u]     + bhh[H_ + u];
  float gg = g[2 * H_ + u] + bhh[2 * H_ + u];
  float go = g[3 * H_ + u] + bhh[3 * H_ + u];
  float i = 1.f / (1.f + __expf(-gi));
  float f = 1.f / (1.f + __expf(-gf));
  float o = 1.f / (1.f + __expf(-go));
  float c2 = f * c[idx] + i * tanhf(gg);
  float h2 = o * tanhf(c2);
  c[idx]  = c2;
  h[idx]  = h2;
  hb[idx] = f2bf(h2);
}

// ---------------------------------------------------------------------------
extern "C" void kernel_launch(void* const* d_in, const int* in_sizes, int n_in,
                              void* d_out, int out_size, void* d_ws, size_t ws_size,
                              hipStream_t stream)
{
  (void)in_sizes; (void)n_in; (void)out_size; (void)ws_size;

  const float* features  = (const float*)d_in[0];
  const int*   captions  = (const int*)  d_in[1];
  const float* embedding = (const float*)d_in[2];
  const float* Wf        = (const float*)d_in[3];
  const float* bf        = (const float*)d_in[4];
  const float* Wh        = (const float*)d_in[5];
  const float* bh        = (const float*)d_in[6];
  const float* Wa        = (const float*)d_in[7];
  const float* ba        = (const float*)d_in[8];
  const float* Wih_att   = (const float*)d_in[9];
  const float* Whh_att   = (const float*)d_in[10];
  const float* bih_att   = (const float*)d_in[11];
  const float* bhh_att   = (const float*)d_in[12];
  const float* Wih_lang  = (const float*)d_in[13];
  const float* Whh_lang  = (const float*)d_in[14];
  const float* bih_lang  = (const float*)d_in[15];
  const float* bhh_lang  = (const float*)d_in[16];
  const float* Wfc       = (const float*)d_in[17];
  const float* bfc       = (const float*)d_in[18];
  float* out = (float*)d_out;

  const int X1 = H_ + F_ + E_;   // 3072
  const int X2 = F_ + H_;        // 2560

  // ---- Workspace layout: f32 region, then bf16 region -----------------------
  float* wsf = (float*)d_ws;
  size_t o = 0;
  float* att1    = wsf + o; o += (size_t)B_ * R_ * A_;
  float* h_att   = wsf + o; o += (size_t)B_ * H_;      // h,c,h,c contiguous
  float* c_att   = wsf + o; o += (size_t)B_ * H_;
  float* h_lang  = wsf + o; o += (size_t)B_ * H_;
  float* c_lang  = wsf + o; o += (size_t)B_ * H_;
  float* att2    = wsf + o; o += (size_t)B_ * A_;
  float* context = wsf + o; o += (size_t)B_ * F_;
  float* gates   = wsf + o; o += (size_t)B_ * 4 * H_;

  __bf16* wsb = (__bf16*)(wsf + o);
  size_t ob = 0;
  __bf16* fb16   = wsb + ob; ob += (size_t)B_ * R_ * F_;   // features
  __bf16* Wfb    = wsb + ob; ob += (size_t)A_ * F_;
  __bf16* Whb    = wsb + ob; ob += (size_t)A_ * H_;
  __bf16* Wihab  = wsb + ob; ob += (size_t)4 * H_ * X1;
  __bf16* Whhab  = wsb + ob; ob += (size_t)4 * H_ * H_;
  __bf16* Wihlb  = wsb + ob; ob += (size_t)4 * H_ * X2;
  __bf16* Whhlb  = wsb + ob; ob += (size_t)4 * H_ * H_;
  __bf16* Wfcb   = wsb + ob; ob += (size_t)V_ * H_;
  __bf16* x1b    = wsb + ob; ob += (size_t)B_ * X1;
  __bf16* x2b    = wsb + ob; ob += (size_t)B_ * X2;
  __bf16* hattb  = wsb + ob; ob += (size_t)B_ * H_;
  __bf16* hlangb = wsb + ob; ob += (size_t)B_ * H_;

  auto cvt = [&](const float* s, __bf16* d, size_t n) {
    int n8 = (int)(n / 8);
    f32_to_bf16_v8<<<(n8 + 255) / 256, 256, 0, stream>>>(s, d, n8);
  };
  // Fused GEMM: C = bias + Xa@Wa^T (+ Xb@Wb^T); ld1 in {512,2048,2560,3072},
  // segment 2 always has leading dim H_.
  auto gemm2 = [&](const __bf16* Xa, int ld1, const __bf16* Wa_, int Ka,
                   const __bf16* Xb, const __bf16* Wb_, int Kb,
                   const float* bias, float* C, int ldc, int M, int N) {
    int waves  = (M / 32) * (N >> 6);
    int blocks = (waves + 7) / 8;            // 8 waves / 256-thread block
    if (ld1 == 512)
      gemm_wmma_bf16<2, 512><<<blocks, 256, 0, stream>>>(
          Xa, Wa_, Ka, Xb, Wb_, Kb, bias, C, ldc, M, N);
    else if (ld1 == 2048)
      gemm_wmma_bf16<2, 2048><<<blocks, 256, 0, stream>>>(
          Xa, Wa_, Ka, Xb, Wb_, Kb, bias, C, ldc, M, N);
    else if (ld1 == 2560)
      gemm_wmma_bf16<2, 2560><<<blocks, 256, 0, stream>>>(
          Xa, Wa_, Ka, Xb, Wb_, Kb, bias, C, ldc, M, N);
    else
      gemm_wmma_bf16<2, 3072><<<blocks, 256, 0, stream>>>(
          Xa, Wa_, Ka, Xb, Wb_, Kb, bias, C, ldc, M, N);
  };

  // ---- One-time per launch: bf16-ify weights + features; zero states --------
  cvt(features, fb16,  (size_t)B_ * R_ * F_);
  cvt(Wf,       Wfb,   (size_t)A_ * F_);
  cvt(Wh,       Whb,   (size_t)A_ * H_);
  cvt(Wih_att,  Wihab, (size_t)4 * H_ * X1);
  cvt(Whh_att,  Whhab, (size_t)4 * H_ * H_);
  cvt(Wih_lang, Wihlb, (size_t)4 * H_ * X2);
  cvt(Whh_lang, Whhlb, (size_t)4 * H_ * H_);
  cvt(Wfc,      Wfcb,  (size_t)V_ * H_);
  {
    int n = 4 * B_ * H_;
    zero_f32<<<(n + 255) / 256, 256, 0, stream>>>(h_att, n);
    int nb = 2 * B_ * H_;   // hattb, hlangb contiguous
    zero_bf16<<<(nb + 255) / 256, 256, 0, stream>>>(hattb, nb);
  }

  // Time-invariant feature projection: att1 = features @ Wf^T + bf
  gemm2(fb16, F_, Wfb, F_, nullptr, nullptr, 0, bf, att1, A_, B_ * R_, A_);

  for (int t = 0; t < T_; ++t) {
    // att2 = h_att @ Wh^T + bh
    gemm2(hattb, H_, Whb, H_, nullptr, nullptr, 0, bh, att2, A_, B_, A_);
    // softmax attention + context (f32)
    attention_kernel<<<B_, 256, 0, stream>>>(att1, att2, Wa, ba, features, context);
    // attention-LSTM: gates = bih + x1@Wih^T + h_att@Whh^T  (register-fused)
    build_x1<<<(B_ * X1 + 255) / 256, 256, 0, stream>>>(h_lang, context,
                                                        embedding, captions, t, x1b);
    gemm2(x1b, X1, Wihab, X1, hattb, Whhab, H_,
          bih_att, gates, 4 * H_, B_, 4 * H_);
    lstm_pointwise<<<(B_ * H_ + 255) / 256, 256, 0, stream>>>(gates, bhh_att,
                                                              h_att, c_att, hattb);
    // language-LSTM: gates = bih + x2@Wih^T + h_lang@Whh^T
    build_x2<<<(B_ * X2 + 255) / 256, 256, 0, stream>>>(context, h_att, x2b);
    gemm2(x2b, X2, Wihlb, X2, hlangb, Whhlb, H_,
          bih_lang, gates, 4 * H_, B_, 4 * H_);
    lstm_pointwise<<<(B_ * H_ + 255) / 256, 256, 0, stream>>>(gates, bhh_lang,
                                                              h_lang, c_lang, hlangb);
    // logits for this step, strided directly into out[B, T, V]
    gemm2(hlangb, H_, Wfcb, H_, nullptr, nullptr, 0,
          bfc, out + (long)t * V_, T_ * V_, B_, V_);
  }
}
